// RNNBaseModel_87608742904506
// MI455X (gfx1250) — matrix operand order repaired and checked by reading, not compiled, not run on hardware
//
#include <hip/hip_runtime.h>

// ---------------------------------------------------------------------------
// Bidirectional LSTM (B=64,T=1024,I=256,H=512,O=128) for MI455X (gfx1250).
// bf16 WMMA everywhere; triple-buffered async global->LDS staging (ASYNCcnt)
// for shared A tiles (one s_barrier per stage); xg in WMMA C-fragment-
// swizzled layout (32B vector I/O); hardware-transcendental LSTM cell update;
// persistent gate-split recurrence with L2-resident weights and per-step
// global barrier.
// ---------------------------------------------------------------------------

#define DEVI __device__ __forceinline__

typedef __bf16 bf16_t;
typedef __attribute__((ext_vector_type(16))) __bf16 v16bf;
typedef __attribute__((ext_vector_type(8)))  __bf16 v8bf;
typedef __attribute__((ext_vector_type(8)))  float  v8f;

static constexpr int B_  = 64;
static constexpr int T_  = 1024;
static constexpr int I_  = 256;
static constexpr int H_  = 512;
static constexpr int O_  = 128;
static constexpr int G_  = 4 * H_;   // 2048 gate columns (i,f,g,o)
static constexpr int H2_ = 2 * H_;   // 1024

union Frag16 { v16bf v; v8bf h[2]; };

// A fragment: 16x32 bf16, row-major source (global or LDS) with leading dim ld.
// lanes 0-15 -> M=lane, K=0..15; lanes 16-31 -> M=lane-16, K=16..31.
DEVI v16bf load_a_frag(const bf16_t* p, size_t ld, int rowBase, int kBase, int lane) {
  size_t r  = (size_t)(rowBase + (lane & 15));
  int    kh = (lane >> 4) * 16;
  const bf16_t* q = p + r * ld + (size_t)(kBase + kh);
  Frag16 f;
  f.h[0] = *(const v8bf*)(q);
  f.h[1] = *(const v8bf*)(q + 8);
  return f.v;
}

// B fragment: 32x16 bf16, weights N-major [col][k] with leading dim ld.
// lanes 0-15 -> N=lane, K=0..15; lanes 16-31 -> N=lane-16, K=16..31.
DEVI v16bf load_b_frag(const bf16_t* p, size_t ld, int colBase, int kBase, int lane) {
  size_t n  = (size_t)(colBase + (lane & 15));
  int    kh = (lane >> 4) * 16;
  return *(const v16bf*)(p + n * ld + (size_t)(kBase + kh));
}

DEVI v8f wmma_bf16(v16bf a, v16bf b, v8f c) {
  return __builtin_amdgcn_wmma_f32_16x16x32_bf16(false, a, false, b, (short)0, c,
                                                 false, false);
}

// Fast activations on the v_exp_f32 / v_rcp_f32 hardware path.
DEVI float fast_sigmoid(float x) {
  return __builtin_amdgcn_rcpf(1.0f + __expf(-x));
}
DEVI float fast_tanh(float x) {
  // tanh(x) = 1 - 2/(exp(2x)+1); exact limits at +-inf.
  return 1.0f - 2.0f * __builtin_amdgcn_rcpf(__expf(2.0f * x) + 1.0f);
}

// --- gfx1250 async global->LDS copy (ASYNCcnt) -----------------------------
DEVI void async_copy16(bf16_t* lds_dst, const bf16_t* gsrc) {
  uint32_t loff = (uint32_t)(uintptr_t)lds_dst;   // low 32 bits = LDS address
  uint64_t ga   = (uint64_t)(uintptr_t)gsrc;
  asm volatile("global_load_async_to_lds_b128 %0, %1, off"
               :: "v"(loff), "v"(ga) : "memory");
}
#define WAIT_ASYNC(n) asm volatile("s_wait_asynccnt " #n ::: "memory")

// Stage a 64-row x 64-col bf16 tile (8KB) into LDS: 512 x 16B chunks,
// 256 threads x 2 chunks each -> each wave issues 2 async instructions.
DEVI void stage_tile_async(bf16_t* tile, const bf16_t* gsrc, size_t row_stride,
                           int tid) {
#pragma unroll
  for (int it = 0; it < 2; ++it) {
    int f   = tid + it * 256;
    int row = f >> 3;
    int cb  = f & 7;
    async_copy16(tile + row * 64 + cb * 8, gsrc + (size_t)row * row_stride + cb * 8);
  }
}

// One 64-wide K stage: 8 ds-A-fragment loads batched ahead of 2x4 WMMAs.
DEVI void stage_compute(const bf16_t* tp, v16bf bf0, v16bf bf1, int lane,
                        v8f acc[4]) {
  v16bf af[4];
#pragma unroll
  for (int m = 0; m < 4; ++m) af[m] = load_a_frag(tp, 64, m * 16, 0, lane);
#pragma unroll
  for (int m = 0; m < 4; ++m) acc[m] = wmma_bf16(af[m], bf0, acc[m]);
#pragma unroll
  for (int m = 0; m < 4; ++m) af[m] = load_a_frag(tp, 64, m * 16, 32, lane);
#pragma unroll
  for (int m = 0; m < 4; ++m) acc[m] = wmma_bf16(af[m], bf1, acc[m]);
}

// ---------------------------------------------------------------------------
// prep kernels
// ---------------------------------------------------------------------------
__global__ void k_cvt_bf16(const float* __restrict__ src, bf16_t* __restrict__ dst,
                           size_t n) {
  size_t i = (size_t)blockIdx.x * blockDim.x + threadIdx.x;
  size_t s = (size_t)gridDim.x * blockDim.x;
  for (; i < n; i += s) dst[i] = (bf16_t)src[i];
}

__global__ void k_bias_sum(const float* __restrict__ a, const float* __restrict__ b,
                           float* __restrict__ o, int n) {
  int i = blockIdx.x * blockDim.x + threadIdx.x;
  if (i < n) o[i] = a[i] + b[i];
}

__global__ void k_zero_bf16(bf16_t* __restrict__ p, size_t n) {
  size_t i = (size_t)blockIdx.x * blockDim.x + threadIdx.x;
  size_t s = (size_t)gridDim.x * blockDim.x;
  for (; i < n; i += s) p[i] = (bf16_t)0.0f;
}

__global__ void k_zero_i32(int* __restrict__ p, int n) {
  int i = blockIdx.x * blockDim.x + threadIdx.x;
  if (i < n) p[i] = 0;
}

// ---------------------------------------------------------------------------
// Phase A: xg = X @ W_ih^T + (b_ih + b_hh), stored fragment-swizzled:
//   xgsw[(((dir*T + t)*128 + nt)*4 + m)*32 + lane][r]  (v8f per lane)
// One WG per (t, 128-gate-col block, dir); M dim = batch (64 rows, 4 tiles).
// ---------------------------------------------------------------------------
__global__ __launch_bounds__(256) void k_gemm_xg(
    const bf16_t* __restrict__ X,    // [B*T][I], row = b*T + t
    const bf16_t* __restrict__ Wih,  // [2][G][I]
    const float*  __restrict__ bias, // [2][G]
    float*        __restrict__ xgsw)
{
  __shared__ __align__(16) bf16_t atile[3][64 * 64];   // 3 x 8KB
  const int t    = blockIdx.x;
  const int nblk = blockIdx.y * 128;
  const int dir  = blockIdx.z;
  const int tid  = (int)threadIdx.x;
  const int wave = tid >> 5, lane = tid & 31;
  const int ncol = nblk + wave * 16;
  const bf16_t* Wd    = Wih + (size_t)dir * G_ * I_;
  const bf16_t* Abase = X + (size_t)t * I_;            // row b at stride T*I
  constexpr int NST = I_ / 64;                          // 4 stages

  stage_tile_async(atile[0], Abase, (size_t)T_ * I_, tid);
  v8f acc[4] = {};
  int bi = 0, bn = 1;
  for (int st = 0; st < NST; ++st) {
    v16bf bf0 = load_b_frag(Wd, I_, ncol, st * 64, lane);
    v16bf bf1 = load_b_frag(Wd, I_, ncol, st * 64 + 32, lane);
    if (st + 1 < NST) {
      stage_tile_async(atile[bn], Abase + (st + 1) * 64, (size_t)T_ * I_, tid);
      WAIT_ASYNC(2);
    } else {
      WAIT_ASYNC(0);
    }
    __syncthreads();
    stage_compute(atile[bi], bf0, bf1, lane, acc);
    bi = bn; bn = (bn == 2) ? 0 : bn + 1;
  }

  const int n  = lane & 15;
  const int nt = ncol >> 4;                      // global gate tile 0..127
  const float bs = bias[(size_t)dir * G_ + ncol + n];
#pragma unroll
  for (int m = 0; m < 4; ++m) {
    v8f v = acc[m];
#pragma unroll
    for (int r = 0; r < 8; ++r) v[r] += bs;
    float* dst = xgsw + (((((size_t)dir * T_ + t) * 128 + nt) * 4 + m) * 32 + lane) * 8;
    *(v8f*)dst = v;                              // 32B store
  }
}

// ---------------------------------------------------------------------------
// Phase B: persistent recurrence. 32 WGs: dir = blk>>4, w = blk&15 owns
// hidden cols [32w,32w+32).
// ---------------------------------------------------------------------------
DEVI void grid_barrier(int* cnt, int members) {
  __syncthreads();
  if (threadIdx.x == 0) {
    __threadfence();
    atomicAdd(cnt, 1);
    volatile int* vc = (volatile int*)cnt;
    while (*vc < members) __builtin_amdgcn_s_sleep(2);
  }
  __syncthreads();
  __threadfence();
}

__global__ __launch_bounds__(256) void k_lstm_rec(
    const float*  __restrict__ xgsw,  // fragment-swizzled xg
    const bf16_t* __restrict__ Whh,   // [2][G][H]
    bf16_t*       __restrict__ hbuf,  // [2][2][B][H] ping-pong
    bf16_t*       __restrict__ Y,     // [B][T][2H]
    float*        __restrict__ hn,    // [2][B][H]
    float*        __restrict__ cn,    // [2][B][H]
    int*          __restrict__ cnt)   // [2][T]
{
  __shared__ __align__(16) bf16_t htile[3][64 * 64];   // 24 KB staging
  __shared__ float gsm[4 * B_ * 32];                   // 32 KB gates
  __shared__ float csm[B_ * 32];                       //  8 KB cell state

  const int dir  = blockIdx.x >> 4;
  const int w    = blockIdx.x & 15;
  const int tid  = (int)threadIdx.x;
  const int wave = tid >> 5;
  const int lane = tid & 31;
  const int gb   = wave >> 1;                 // gate block 0..3 (i,f,g,o)
  const int nh   = wave & 1;
  const int gcolBase = gb * H_ + w * 32 + nh * 16;
  const int ntG  = gcolBase >> 4;             // global gate tile of this wave
  const bool rev = (dir == 1);

  // vectorized cell-update mapping: thread owns (batch row eb, 8-col quarter)
  const int eb = tid >> 2;
  const int eq = (tid & 3) * 8;

  for (int e = tid; e < B_ * 32; e += 256) csm[e] = 0.0f;
  __syncthreads();

  const bf16_t* WhhD = Whh + (size_t)dir * G_ * H_;
  const int n    = lane & 15;
  const int rofs = (lane >> 4) * 8;
  constexpr int NST = H_ / 64;                 // 8 stages

  for (int s = 0; s < T_; ++s) {
    const int t = rev ? (T_ - 1 - s) : s;
    const bf16_t* hprev = hbuf + ((size_t)dir * 2 + (s & 1)) * (B_ * H_);
    bf16_t*       hnext = hbuf + ((size_t)dir * 2 + ((s & 1) ^ 1)) * (B_ * H_);

    stage_tile_async(htile[0], hprev, H_, tid);
    v8f acc[4] = {};
    int bi = 0, bn = 1;
    for (int st = 0; st < NST; ++st) {
      v16bf bf0 = load_b_frag(WhhD, H_, gcolBase, st * 64, lane);
      v16bf bf1 = load_b_frag(WhhD, H_, gcolBase, st * 64 + 32, lane);
      if (st + 1 < NST) {
        stage_tile_async(htile[bn], hprev + (st + 1) * 64, H_, tid);
        __builtin_prefetch(WhhD + (size_t)(gcolBase + n) * H_ + (st + 1) * 64, 0, 1);
        WAIT_ASYNC(2);
      } else {
        WAIT_ASYNC(0);
      }
      __syncthreads();
      stage_compute(htile[bi], bf0, bf1, lane, acc);
      bi = bn; bn = (bn == 2) ? 0 : bn + 1;
    }

    // add precomputed input gates (32B vector loads), stash to LDS
    const float* xgt = xgsw + ((((size_t)dir * T_ + t) * 128 + ntG) * 4) * 256;
#pragma unroll
    for (int m = 0; m < 4; ++m) {
      v8f xv = *(const v8f*)(xgt + (size_t)m * 256 + lane * 8);
      v8f g  = acc[m] + xv;
#pragma unroll
      for (int r = 0; r < 8; ++r) {
        int b = m * 16 + r + rofs;
        gsm[(gb * B_ + b) * 32 + nh * 16 + n] = g[r];
      }
    }
    __syncthreads();

    // vectorized LSTM cell update: thread -> (row eb, cols [eq, eq+8))
    {
      v8f iv = *(const v8f*)&gsm[(0 * B_ + eb) * 32 + eq];
      v8f fv = *(const v8f*)&gsm[(1 * B_ + eb) * 32 + eq];
      v8f gv = *(const v8f*)&gsm[(2 * B_ + eb) * 32 + eq];
      v8f ov = *(const v8f*)&gsm[(3 * B_ + eb) * 32 + eq];
      v8f cv = *(const v8f*)&csm[eb * 32 + eq];
      v8f hv;
      v8bf hb;
#pragma unroll
      for (int r = 0; r < 8; ++r) {
        float ig = fast_sigmoid(iv[r]);
        float fg = fast_sigmoid(fv[r]);
        float gg = fast_tanh(gv[r]);
        float og = fast_sigmoid(ov[r]);
        float c  = fg * cv[r] + ig * gg;
        cv[r] = c;
        float h = og * fast_tanh(c);
        hv[r] = h;
        hb[r] = (bf16_t)h;
      }
      *(v8f*)&csm[eb * 32 + eq] = cv;
      const int hc = w * 32 + eq;
      *(v8bf*)&hnext[(size_t)eb * H_ + hc] = hb;                          // 16B
      *(v8bf*)&Y[((size_t)eb * T_ + t) * H2_ + (size_t)dir * H_ + hc] = hb;
      if (s == T_ - 1) {
        *(v8f*)&hn[((size_t)dir * B_ + eb) * H_ + hc] = hv;               // 32B
        *(v8f*)&cn[((size_t)dir * B_ + eb) * H_ + hc] = cv;
      }
    }
    grid_barrier(cnt + dir * T_ + s, 16);
  }
}

// ---------------------------------------------------------------------------
// Phase C: out = Y @ W_out^T + b_out. Y tile async-staged to LDS.
// ---------------------------------------------------------------------------
__global__ __launch_bounds__(256) void k_gemm_out(
    const bf16_t* __restrict__ Y,     // [B*T][2H]
    const bf16_t* __restrict__ Wo,    // [O][2H]
    const float*  __restrict__ bout,  // [O]
    float*        __restrict__ out)   // [B*T][O]
{
  __shared__ __align__(16) bf16_t atile[3][64 * 64];
  const int mblk = blockIdx.x * 64;
  const int tid  = (int)threadIdx.x;
  const int wave = tid >> 5, lane = tid & 31;
  const int ncol = wave * 16;
  const bf16_t* Abase = Y + (size_t)mblk * H2_;
  constexpr int NST = H2_ / 64;                // 16 stages

  stage_tile_async(atile[0], Abase, H2_, tid);
  v8f acc[4] = {};
  int bi = 0, bn = 1;
  for (int st = 0; st < NST; ++st) {
    v16bf bf0 = load_b_frag(Wo, H2_, ncol, st * 64, lane);
    v16bf bf1 = load_b_frag(Wo, H2_, ncol, st * 64 + 32, lane);
    if (st + 1 < NST) {
      stage_tile_async(atile[bn], Abase + (st + 1) * 64, H2_, tid);
      WAIT_ASYNC(2);
    } else {
      WAIT_ASYNC(0);
    }
    __syncthreads();
    stage_compute(atile[bi], bf0, bf1, lane, acc);
    bi = bn; bn = (bn == 2) ? 0 : bn + 1;
  }

  const int n    = lane & 15;
  const int rofs = (lane >> 4) * 8;
  const float bb = bout[ncol + n];
#pragma unroll
  for (int m = 0; m < 4; ++m)
#pragma unroll
    for (int r = 0; r < 8; ++r) {
      size_t row = (size_t)(mblk + m * 16 + r + rofs);
      out[row * O_ + ncol + n] = acc[m][r] + bb;
    }
}

// ---------------------------------------------------------------------------
// launcher
// ---------------------------------------------------------------------------
extern "C" void kernel_launch(void* const* d_in, const int* in_sizes, int n_in,
                              void* d_out, int out_size, void* d_ws, size_t ws_size,
                              hipStream_t stream) {
  (void)in_sizes; (void)n_in; (void)out_size; (void)ws_size;

  const float* x    = (const float*)d_in[0];
  const float* wihf = (const float*)d_in[1];
  const float* whhf = (const float*)d_in[2];
  const float* bihf = (const float*)d_in[3];
  const float* bhhf = (const float*)d_in[4];
  const float* wihb = (const float*)d_in[5];
  const float* whhb = (const float*)d_in[6];
  const float* bihb = (const float*)d_in[7];
  const float* bhhb = (const float*)d_in[8];
  const float* wout = (const float*)d_in[9];
  const float* bout = (const float*)d_in[10];

  float* out = (float*)d_out;
  float* hn  = out + (size_t)B_ * T_ * O_;
  float* cn  = hn + (size_t)2 * B_ * H_;

  char*  ws  = (char*)d_ws;
  size_t off = 0;
  auto carve = [&](size_t bytes) -> void* {
    void* p = ws + off;
    off = (off + bytes + 255) & ~(size_t)255;
    return p;
  };
  bf16_t* Xbf  = (bf16_t*)carve((size_t)B_ * T_ * I_ * 2);
  bf16_t* Wih  = (bf16_t*)carve((size_t)2 * G_ * I_ * 2);
  bf16_t* Whh  = (bf16_t*)carve((size_t)2 * G_ * H_ * 2);
  bf16_t* Wo   = (bf16_t*)carve((size_t)O_ * H2_ * 2);
  float*  bias = (float*) carve((size_t)2 * G_ * 4);
  bf16_t* hbuf = (bf16_t*)carve((size_t)2 * 2 * B_ * H_ * 2);
  int*    cnt  = (int*)   carve((size_t)2 * T_ * 4);
  bf16_t* Y    = (bf16_t*)carve((size_t)B_ * T_ * H2_ * 2);
  float*  xgsw = (float*) carve((size_t)2 * T_ * B_ * G_ * 4);

  k_cvt_bf16<<<2048, 256, 0, stream>>>(x,    Xbf, (size_t)B_ * T_ * I_);
  k_cvt_bf16<<<512,  256, 0, stream>>>(wihf, Wih,                   (size_t)G_ * I_);
  k_cvt_bf16<<<512,  256, 0, stream>>>(wihb, Wih + (size_t)G_ * I_, (size_t)G_ * I_);
  k_cvt_bf16<<<512,  256, 0, stream>>>(whhf, Whh,                   (size_t)G_ * H_);
  k_cvt_bf16<<<512,  256, 0, stream>>>(whhb, Whh + (size_t)G_ * H_, (size_t)G_ * H_);
  k_cvt_bf16<<<128,  256, 0, stream>>>(wout, Wo,                    (size_t)O_ * H2_);
  k_bias_sum<<<8, 256, 0, stream>>>(bihf, bhhf, bias,      G_);
  k_bias_sum<<<8, 256, 0, stream>>>(bihb, bhhb, bias + G_, G_);
  k_zero_bf16<<<64, 256, 0, stream>>>(hbuf, (size_t)2 * 2 * B_ * H_);
  k_zero_i32<<<8, 256, 0, stream>>>(cnt, 2 * T_);

  // phase A: input-gate GEMM, one WG per (t, 128 gate cols, dir)
  k_gemm_xg<<<dim3(T_, G_ / 128, 2), 256, 0, stream>>>(Xbf, Wih, bias, xgsw);

  // phase B: persistent bidirectional recurrence (16 WGs per direction)
  k_lstm_rec<<<32, 256, 0, stream>>>(xgsw, Whh, hbuf, Y, hn, cn, cnt);

  // phase C: output projection
  k_gemm_out<<<(B_ * T_) / 64, 256, 0, stream>>>(Y, Wo, bout, out);
}
